// Attention_41850161332757
// MI455X (gfx1250) — matrix operand order repaired and checked by reading, not compiled, not run on hardware
//
#include <hip/hip_runtime.h>
#include <hip/hip_bf16.h>
#include <math.h>

// ---------------- types ----------------
typedef __bf16 bhalf;
typedef __attribute__((ext_vector_type(16))) __bf16 v16bf;
typedef __attribute__((ext_vector_type(8)))  __bf16 v8bf;
typedef __attribute__((ext_vector_type(8)))  float  v8f;

// ---------------- problem constants ----------------
#define BB 2
#define HH 8
#define DD 32
#define QQ 2048
#define KL 2048            // K sequence length
#define CC 256
#define HD 256             // H*D
#define LOG2E 1.4426950408889634f
#define RSQRT_D 0.17677669529663687f   // 1/sqrt(32)

// ---------------- WMMA helpers ----------------
__device__ __forceinline__ v8f wmma_bf16f32(v16bf a, v16bf b, v8f c) {
  // D = A(16x32 bf16) * B(32x16 bf16) + C(16x16 f32)
  return __builtin_amdgcn_wmma_f32_16x16x32_bf16(
      /*neg_a=*/false, a, /*neg_b=*/false, b,
      /*c_mod=*/(short)0, c, /*reuse_a=*/false, /*reuse_b=*/false);
}

// A-matrix 16x32 bf16 fragment: lane<16 -> row=lane, K={0..15};
// lane>=16 -> row=lane-16, K={16..31}. p points at tile origin, row-major ld.
__device__ __forceinline__ v16bf load_a_frag(const bhalf* p, int ld, int lane) {
  const int row = lane & 15, hi = lane >> 4;
  const v8bf lo = *reinterpret_cast<const v8bf*>(p + (size_t)row * ld + 16 * hi);
  const v8bf hv = *reinterpret_cast<const v8bf*>(p + (size_t)row * ld + 16 * hi + 8);
  v16bf a;
#pragma unroll
  for (int i = 0; i < 8; i++) { a[i] = lo[i]; a[i + 8] = hv[i]; }
  return a;
}

// B-matrix 32x16 bf16 fragment: lane holds column n=lane&15, K = 16*hi + {0..15}.
// Source stored as [n][k] row-major with leading dim ld (contiguous 32B per lane).
__device__ __forceinline__ v16bf load_b_frag(const bhalf* p, int ld, int lane) {
  const int nn = lane & 15, hi = lane >> 4;
  return *reinterpret_cast<const v16bf*>(p + (size_t)nn * ld + 16 * hi);
}

__device__ __forceinline__ float rmax16(float v) {
#pragma unroll
  for (int m = 1; m < 16; m <<= 1) v = fmaxf(v, __shfl_xor(v, m, 32));
  return v;
}
__device__ __forceinline__ float rsum16(float v) {
#pragma unroll
  for (int m = 1; m < 16; m <<= 1) v += __shfl_xor(v, m, 32);
  return v;
}

// Async DMA: copy 16B from global to LDS, tracked by ASYNCcnt (gfx1250).
// VDST VGPR carries the LDS byte address (ISA §10 / §15.18.3, opcode 98).
__device__ __forceinline__ void async_load_b128(unsigned lds_addr, const void* gptr) {
  asm volatile("global_load_async_to_lds_b128 %0, %1, off"
               :: "v"(lds_addr), "v"(gptr)
               : "memory");
}
__device__ __forceinline__ void wait_asynccnt0() {
  asm volatile("s_wait_asynccnt 0x0" ::: "memory");
}

// ---------------- kernel 0: f32 -> bf16 ----------------
__global__ void cvt_f32_bf16(const float* __restrict__ src, bhalf* __restrict__ dst, int n) {
  const int i = blockIdx.x * blockDim.x + threadIdx.x;
  if (i < n) dst[i] = (bhalf)src[i];
}

// ---------------- kernel 1: fused Q/K/V/G projections ----------------
// grid.x = 128 (8 waves/block, 1024 waves), grid.y = 4 (proj id)
// each wave: 16 rows x 64 cols output tile, K-loop over C=256 in steps of 32.
__global__ __launch_bounds__(256) void proj_kernel(
    const bhalf* __restrict__ qx, const bhalf* __restrict__ kvx,
    const bhalf* __restrict__ Wq, const bhalf* __restrict__ Wk,
    const bhalf* __restrict__ Wv, const bhalf* __restrict__ Wg,
    const float* __restrict__ bg,
    bhalf* __restrict__ qh,   // [B,H,Q,D] bf16, pre-scaled by 1/sqrt(D)
    bhalf* __restrict__ kh,   // [B,H,K,D] bf16
    bhalf* __restrict__ vt,   // [B,H,D,K] bf16 (transposed V)
    float* __restrict__ gate) // [B,Q,HD] f32 = sigmoid(qx@Wg^T + bg)
{
  const int lane = threadIdx.x & 31;
  const int w = blockIdx.x * 8 + (threadIdx.x >> 5);
  const int proj = blockIdx.y;
  const int mt = w >> 2, nc = w & 3;
  const int m0 = mt * 16, n0 = nc * 64;
  const int n = lane & 15, hi = lane >> 4;

  const bhalf* X = (proj == 1 || proj == 2) ? kvx : qx;
  const bhalf* W = (proj == 0) ? Wq : (proj == 1) ? Wk : (proj == 2) ? Wv : Wg;

  v8f acc[4];
#pragma unroll
  for (int t = 0; t < 4; t++)
#pragma unroll
    for (int r = 0; r < 8; r++) acc[t][r] = 0.f;

#pragma unroll
  for (int kc = 0; kc < CC; kc += 32) {
    v16bf a = load_a_frag(X + (size_t)m0 * CC + kc, CC, lane);
#pragma unroll
    for (int t = 0; t < 4; t++) {
      v16bf bw = load_b_frag(W + (size_t)(n0 + t * 16) * CC + kc, CC, lane);
      acc[t] = wmma_bf16f32(a, bw, acc[t]);
    }
  }

  const int b  = m0 / QQ;
  const int r0 = m0 % QQ;
#pragma unroll
  for (int t = 0; t < 4; t++) {
    const int col = n0 + t * 16 + n;
    const int h = col >> 5, d = col & 31;
    const float bgc = (proj == 3) ? bg[col] : 0.f;
#pragma unroll
    for (int r = 0; r < 8; r++) {
      const int row = r0 + r + 8 * hi;   // q or k index within batch
      const float v = acc[t][r];
      if (proj == 0) {
        qh[((size_t)(b * HH + h) * QQ + row) * DD + d] = (bhalf)(v * RSQRT_D);
      } else if (proj == 1) {
        kh[((size_t)(b * HH + h) * KL + row) * DD + d] = (bhalf)v;
      } else if (proj == 2) {
        vt[((size_t)(b * HH + h) * DD + d) * KL + row] = (bhalf)v;
      } else {
        const float s = v + bgc;
        gate[(size_t)(m0 + r + 8 * hi) * HD + col] = 1.0f / (1.0f + __expf(-s));
      }
    }
  }
}

// ---------------- kernel 2: flash attention ----------------
// Block = 8 waves sharing one (b,h); wave wv handles q-tile (qc*8 + wv).
// K/V tiles for each 64-wide K block are staged once per block into LDS via
// GLOBAL_LOAD_ASYNC_TO_LDS_B128 (each wave DMAs 512B of K and 512B of V).
__global__ __launch_bounds__(256) void attn_kernel(
    const bhalf* __restrict__ qh, const bhalf* __restrict__ kh,
    const bhalf* __restrict__ vt,
    const float* __restrict__ bias_mask,  // [B,K]
    const float* __restrict__ bias_pair,  // [H,Q,K]
    const float* __restrict__ gate,       // [B,Q,HD]
    bhalf* __restrict__ op)               // [B,Q,HD] bf16 = (softmax@V)*gate
{
  __shared__ __align__(32) bhalf ldsK[64 * 32];   // K tile  [krow][d]
  __shared__ __align__(32) bhalf ldsV[32 * 64];   // V^T tile [d][k]
  __shared__ __align__(32) bhalf ldsP[8][16 * 72]; // per-wave P tile (padded)

  const int lane = threadIdx.x & 31, wv = threadIdx.x >> 5;
  const int bh = blockIdx.x >> 4;          // 16 (b,h) pairs
  const int qc = blockIdx.x & 15;          // 16 q-chunks of 8 tiles
  const int b = bh >> 3, h = bh & 7;
  const int n = lane & 15, hi = lane >> 4;
  const int qt = qc * 8 + wv;              // this wave's q-tile
  const int q0 = qt * 16;

  const bhalf* qh_b = qh + (size_t)(b * HH + h) * QQ * DD + (size_t)q0 * DD;
  const bhalf* kh_b = kh + (size_t)(b * HH + h) * KL * DD;
  const bhalf* vt_b = vt + (size_t)(b * HH + h) * DD * KL;
  const float* bp_h = bias_pair + ((size_t)h * QQ + q0) * KL;
  const float* bm_b = bias_mask + (size_t)b * KL;

  const v16bf aq = load_a_frag(qh_b, DD, lane);  // Q tile, reused across K loop

  float mrow[8], lrow[8];
  v8f o0, o1;
#pragma unroll
  for (int r = 0; r < 8; r++) { mrow[r] = -__builtin_inff(); lrow[r] = 0.f; o0[r] = 0.f; o1[r] = 0.f; }

  const unsigned ldsK_base = (unsigned)(uintptr_t)&ldsK[0];
  const unsigned ldsV_base = (unsigned)(uintptr_t)&ldsV[0];

  for (int k0 = 0; k0 < KL; k0 += 64) {
    __syncthreads();   // previous iteration's LDS reads done before overwrite

    // ---- stage K tile (4KB contiguous) and V^T tile via async DMA ----
    {
      // K: byte offset wv*512 + lane*16 within the contiguous 64x32 tile
      const unsigned koff = (unsigned)(wv * 512 + lane * 16);
      async_load_b128(ldsK_base + koff,
                      (const char*)(kh_b + (size_t)k0 * DD) + koff);
      // V^T: wave wv covers d rows 4*wv..4*wv+3, each row 128B in 16B chunks
      const int d = wv * 4 + (lane >> 3);
      const unsigned coff = (unsigned)((lane & 7) * 16);
      async_load_b128(ldsV_base + (unsigned)(d * 128) + coff,
                      (const char*)(vt_b + (size_t)d * KL + k0) + coff);
    }

    // ---- bias_mask + bias_pair into WMMA C operands (overlaps the DMA) ----
    v8f cbias[4];
#pragma unroll
    for (int t = 0; t < 4; t++) {
      const int kcol = k0 + t * 16 + n;
      const float bm = bm_b[kcol];
      const float* bp = bp_h + kcol;
#pragma unroll
      for (int r = 0; r < 8; r++) cbias[t][r] = bp[(size_t)(r + 8 * hi) * KL] + bm;
    }
    if (k0 + 64 < KL) {
#pragma unroll
      for (int r = 0; r < 8; r++)
        __builtin_prefetch(bp_h + (size_t)(r + 8 * hi) * KL + k0 + 64 + n, 0, 3);
    }

    wait_asynccnt0();   // own DMAs landed in LDS
    __syncthreads();    // all waves' DMAs visible

    // ---- S = Q K^T + bias (K fragments from LDS) ----
    v8f s[4];
#pragma unroll
    for (int t = 0; t < 4; t++) {
      v16bf bk = load_b_frag(&ldsK[t * 16 * 32], 32, lane);
      s[t] = wmma_bf16f32(aq, bk, cbias[t]);
    }

    // ---- online softmax: stats per row r+8*hi live exactly in this lane ----
#pragma unroll
    for (int r = 0; r < 8; r++) {
      float tmax = fmaxf(fmaxf(s[0][r], s[1][r]), fmaxf(s[2][r], s[3][r]));
      tmax = rmax16(tmax);
      const float mnew = fmaxf(mrow[r], tmax);
      const float scl = exp2f((mrow[r] - mnew) * LOG2E);
      mrow[r] = mnew;
      float rsum = 0.f;
#pragma unroll
      for (int t = 0; t < 4; t++) {
        const float p = exp2f((s[t][r] - mnew) * LOG2E);
        s[t][r] = p;
        rsum += p;
      }
      rsum = rsum16(rsum);
      lrow[r] = lrow[r] * scl + rsum;
      o0[r] *= scl;
      o1[r] *= scl;
    }

    // ---- transpose P (D-layout -> A-layout) through private LDS tile ----
    bhalf* myP = &ldsP[wv][0];
#pragma unroll
    for (int t = 0; t < 4; t++)
#pragma unroll
      for (int r = 0; r < 8; r++)
        myP[(r + 8 * hi) * 72 + t * 16 + n] = (bhalf)s[t][r];

    // ---- O += P @ V  (V^T fragments from LDS) ----
#pragma unroll
    for (int kc = 0; kc < 2; kc++) {
      v16bf ap  = load_a_frag(myP + kc * 32, 72, lane);
      v16bf bv0 = load_b_frag(&ldsV[kc * 32], 64, lane);
      v16bf bv1 = load_b_frag(&ldsV[16 * 64 + kc * 32], 64, lane);
      o0 = wmma_bf16f32(ap, bv0, o0);
      o1 = wmma_bf16f32(ap, bv1, o1);
    }
  }

  // ---- epilogue: normalize, gate, store bf16 ----
#pragma unroll
  for (int r = 0; r < 8; r++) {
    const int q = q0 + r + 8 * hi;
    const float inv = 1.0f / lrow[r];
    const size_t gbase = (size_t)(b * QQ + q) * HD + h * DD;
    const float g0 = gate[gbase + n];
    const float g1 = gate[gbase + 16 + n];
    op[gbase + n]      = (bhalf)(o0[r] * inv * g0);
    op[gbase + 16 + n] = (bhalf)(o1[r] * inv * g1);
  }
}

// ---------------- kernel 3: output projection out = op @ Wo^T + bo ----------------
__global__ __launch_bounds__(256) void out_kernel(
    const bhalf* __restrict__ op, const bhalf* __restrict__ Wo,
    const float* __restrict__ bo, float* __restrict__ out)
{
  const int lane = threadIdx.x & 31;
  const int w = blockIdx.x * 8 + (threadIdx.x >> 5);
  const int mt = w >> 2, nc = w & 3;
  const int m0 = mt * 16, n0 = nc * 64;
  const int n = lane & 15, hi = lane >> 4;

  v8f acc[4];
#pragma unroll
  for (int t = 0; t < 4; t++)
#pragma unroll
    for (int r = 0; r < 8; r++) acc[t][r] = 0.f;

#pragma unroll
  for (int kc = 0; kc < HD; kc += 32) {
    v16bf a = load_a_frag(op + (size_t)m0 * HD + kc, HD, lane);
#pragma unroll
    for (int t = 0; t < 4; t++) {
      v16bf bw = load_b_frag(Wo + (size_t)(n0 + t * 16) * HD + kc, HD, lane);
      acc[t] = wmma_bf16f32(a, bw, acc[t]);
    }
  }

#pragma unroll
  for (int t = 0; t < 4; t++) {
    const int col = n0 + t * 16 + n;
    const float bias = bo[col];
#pragma unroll
    for (int r = 0; r < 8; r++)
      out[(size_t)(m0 + r + 8 * hi) * CC + col] = acc[t][r] + bias;
  }
}

// ---------------- host launcher ----------------
extern "C" void kernel_launch(void* const* d_in, const int* in_sizes, int n_in,
                              void* d_out, int out_size, void* d_ws, size_t ws_size,
                              hipStream_t stream) {
  const float* q_x       = (const float*)d_in[0];
  const float* kv_x      = (const float*)d_in[1];
  const float* bias_mask = (const float*)d_in[2];
  const float* bias_pair = (const float*)d_in[3];
  const float* Wq        = (const float*)d_in[4];
  const float* Wk        = (const float*)d_in[5];
  const float* Wv        = (const float*)d_in[6];
  const float* Wo        = (const float*)d_in[7];
  const float* bo        = (const float*)d_in[8];
  const float* Wg        = (const float*)d_in[9];
  const float* bg        = (const float*)d_in[10];
  float* out = (float*)d_out;

  char* ws = (char*)d_ws;
  size_t off = 0;
  auto take = [&](size_t bytes) -> char* {
    char* p = ws + off;
    off += (bytes + 255) & ~(size_t)255;
    return p;
  };
  bhalf* qx_b  = (bhalf*)take((size_t)BB * QQ * CC * 2);
  bhalf* kvx_b = (bhalf*)take((size_t)BB * KL * CC * 2);
  bhalf* Wq_b  = (bhalf*)take((size_t)HD * CC * 2);
  bhalf* Wk_b  = (bhalf*)take((size_t)HD * CC * 2);
  bhalf* Wv_b  = (bhalf*)take((size_t)HD * CC * 2);
  bhalf* Wg_b  = (bhalf*)take((size_t)HD * CC * 2);
  bhalf* Wo_b  = (bhalf*)take((size_t)CC * HD * 2);
  bhalf* qh    = (bhalf*)take((size_t)BB * HH * QQ * DD * 2);
  bhalf* kh    = (bhalf*)take((size_t)BB * HH * KL * DD * 2);
  bhalf* vt    = (bhalf*)take((size_t)BB * HH * DD * KL * 2);
  float* gate  = (float*)take((size_t)BB * QQ * HD * 4);
  bhalf* op    = (bhalf*)take((size_t)BB * QQ * HD * 2);

  auto cvt = [&](const float* s, bhalf* d, int nel) {
    cvt_f32_bf16<<<(nel + 255) / 256, 256, 0, stream>>>(s, d, nel);
  };
  cvt(q_x,  qx_b,  BB * QQ * CC);
  cvt(kv_x, kvx_b, BB * KL * CC);
  cvt(Wq, Wq_b, HD * CC);
  cvt(Wk, Wk_b, HD * CC);
  cvt(Wv, Wv_b, HD * CC);
  cvt(Wg, Wg_b, HD * CC);
  cvt(Wo, Wo_b, CC * HD);

  proj_kernel<<<dim3(128, 4), 256, 0, stream>>>(qx_b, kvx_b, Wq_b, Wk_b, Wv_b, Wg_b,
                                                bg, qh, kh, vt, gate);
  attn_kernel<<<256, 256, 0, stream>>>(qh, kh, vt, bias_mask, bias_pair, gate, op);
  out_kernel<<<128, 256, 0, stream>>>(op, Wo_b, bo, out);
}
